// GQA_57818849739407
// MI455X (gfx1250) — matrix operand order repaired and checked by reading, not compiled, not run on hardware
//
#include <hip/hip_runtime.h>
#include <hip/hip_bf16.h>
#include <math.h>

// ---------------- CDNA5 WMMA types ----------------
typedef __attribute__((ext_vector_type(16))) __bf16 v16bf;
typedef __attribute__((ext_vector_type(8)))  __bf16 v8bf;
typedef __attribute__((ext_vector_type(8)))  float  v8f;

#define B_   2
#define T_   2048
#define C_   4096
#define H_   32
#define HKV_ 8
#define DH_  128

// LDS row strides (elements) with +8 bf16 (16B) padding to break bank conflicts
#define LDAB_ 72    // gemm tiles: 64-elem rows
#define LDK_  136   // attn K tile: 128-elem rows
#define LDV_  40    // attn V tile: 32-elem rows
#define LDP_  40    // attn P tile: 32-elem rows

// scheduling hints: pin "N DS-reads then M WMMAs" bursts (no-op if unsupported)
#if __has_builtin(__builtin_amdgcn_sched_group_barrier)
#define SGB(mask, n, id) __builtin_amdgcn_sched_group_barrier((mask), (n), (id))
#else
#define SGB(mask, n, id)
#endif

// float -> bf16, round to nearest even
__device__ __forceinline__ unsigned short f2bf(float f){
  unsigned int u = __float_as_uint(f);
  u += 0x7FFFu + ((u >> 16) & 1u);
  return (unsigned short)(u >> 16);
}

__device__ __forceinline__ v8f wmma_bf16(v16bf a, v16bf b, v8f c){
  // D = A(16x32 bf16) * B(32x16 bf16) + C(16x16 f32)
  return __builtin_amdgcn_wmma_f32_16x16x32_bf16(false, a, false, b, (short)0, c, false, false);
}

// A-fragment: 16 contiguous bf16 at p (K koff..koff+15)
__device__ __forceinline__ v16bf load_afrag(const unsigned short* p){
  v8bf lo = *(const v8bf*)p;
  v8bf hi = *(const v8bf*)(p + 8);
  v16bf f;
#pragma unroll
  for (int i = 0; i < 8; i++){ f[i] = lo[i]; f[i + 8] = hi[i]; }
  return f;
}

// B-fragment: 16 contiguous bf16 (one column, K consecutive); 16B-aligned halves
__device__ __forceinline__ v16bf load_bfrag(const unsigned short* p){
  v8bf lo = *(const v8bf*)p;
  v8bf hi = *(const v8bf*)(p + 8);
  v16bf f;
#pragma unroll
  for (int i = 0; i < 8; i++){ f[i] = lo[i]; f[i + 8] = hi[i]; }
  return f;
}

// butterfly reductions within each 16-lane half of the wave32
__device__ __forceinline__ float rmax16(float v){
#pragma unroll
  for (int m = 1; m < 16; m <<= 1) v = fmaxf(v, __shfl_xor(v, m, 32));
  return v;
}
__device__ __forceinline__ float rsum16(float v){
#pragma unroll
  for (int m = 1; m < 16; m <<= 1) v += __shfl_xor(v, m, 32);
  return v;
}

// ---------------- elementwise f32 -> bf16 (x4) ----------------
__global__ __launch_bounds__(256) void cvt_bf16x4_kernel(const float* __restrict__ in,
                                                         unsigned short* __restrict__ out, int n4){
  int i = blockIdx.x * 256 + threadIdx.x;
  if (i >= n4) return;
  float4 v = ((const float4*)in)[i];
  union { unsigned short s[4]; uint2 u; } o;
  o.s[0] = f2bf(v.x); o.s[1] = f2bf(v.y); o.s[2] = f2bf(v.z); o.s[3] = f2bf(v.w);
  ((uint2*)out)[i] = o.u;
}

// ---------------- f32 [K][N] -> bf16 [N][K] (weight transpose) ----------------
__global__ __launch_bounds__(256) void transpose_cvt_kernel(const float* __restrict__ in,
                                                            unsigned short* __restrict__ out,
                                                            int K, int N){
  __shared__ float tile[32][33];
  const int n0 = blockIdx.x * 32, k0 = blockIdx.y * 32;
  const int tx = threadIdx.x, ty = threadIdx.y; // block (32,8)
#pragma unroll
  for (int r = ty; r < 32; r += 8)
    tile[r][tx] = in[(size_t)(k0 + r) * N + n0 + tx];
  __syncthreads();
#pragma unroll
  for (int r = ty; r < 32; r += 8)
    out[(size_t)(n0 + r) * K + k0 + tx] = f2bf(tile[tx][r]);
}

// ------------- shared GEMM core: 128x128 tile, K-step 64, 512 thr / 16 waves -------------
// A: bf16 [M][K] row-major.  Bt: bf16 [N][K] (B pre-transposed).
// Wave w: m-tile (w>>1) rows [16(w>>1),+16); N-tiles interleaved by parity p=w&1:
// acc[j] covers n-tile jj = p + 2j  (so RoPE pairs (jj, jj+4) share a parity/wave).
// Staging values are SCALAR uint4s (never an alloca -> stay in VGPRs).
__device__ __forceinline__ void gemm_core(const unsigned short* __restrict__ A,
                                          const unsigned short* __restrict__ Bt,
                                          int K, size_t m0, size_t n0,
                                          unsigned short* lA, unsigned short* lB,
                                          v8f (&acc)[4]){
  const int tid  = threadIdx.x;
  const int wave = tid >> 5, lane = tid & 31;
  const int ncol = lane & 15, seg = lane >> 4, koff = seg * 16;
  const int mt = wave >> 1, p = wave & 1;
  const int row = tid >> 2, q = tid & 3;   // each thread stages 32B of A and 32B of B
  const v8f z = {0.f,0.f,0.f,0.f,0.f,0.f,0.f,0.f};
#pragma unroll
  for (int j = 0; j < 4; j++) acc[j] = z;

  const unsigned short* gA = A  + (m0 + row) * (size_t)K + q * 16;
  const unsigned short* gB = Bt + (n0 + row) * (size_t)K + q * 16;
  unsigned short* sA = lA + row * LDAB_ + q * 16;
  unsigned short* sB = lB + row * LDAB_ + q * 16;
  const unsigned short* fA = lA + (mt * 16 + ncol) * LDAB_ + koff;

  uint4 ra0 = *(const uint4*)(gA);
  uint4 ra1 = *(const uint4*)(gA + 8);
  uint4 rb0 = *(const uint4*)(gB);
  uint4 rb1 = *(const uint4*)(gB + 8);

  for (int k0 = 0; k0 < K; k0 += 64){
    // commit staged registers to LDS
    *(uint4*)(sA)     = ra0;  *(uint4*)(sA + 8) = ra1;
    *(uint4*)(sB)     = rb0;  *(uint4*)(sB + 8) = rb1;
    __syncthreads();
    // fetch next tile while the WMMA chain runs
    if (k0 + 64 < K){
      ra0 = *(const uint4*)(gA + k0 + 64);
      ra1 = *(const uint4*)(gA + k0 + 72);
      rb0 = *(const uint4*)(gB + k0 + 64);
      rb1 = *(const uint4*)(gB + k0 + 72);
      if (k0 + 128 < K){
        __builtin_prefetch(gA + k0 + 128, 0, 3);   // global_prefetch_b8 (SE scope)
        __builtin_prefetch(gB + k0 + 128, 0, 3);
      }
    }
    // two 16x16x32 sub-steps; group all 10 DS reads, then 4 back-to-back WMMAs
#pragma unroll
    for (int c = 0; c < 2; c++){
      v16bf af = load_afrag(fA + c * 32);
      v16bf bf0 = load_bfrag(lB + ((p     ) * 16 + ncol) * LDAB_ + c * 32 + seg * 16);
      v16bf bf1 = load_bfrag(lB + ((p + 2) * 16 + ncol) * LDAB_ + c * 32 + seg * 16);
      v16bf bf2 = load_bfrag(lB + ((p + 4) * 16 + ncol) * LDAB_ + c * 32 + seg * 16);
      v16bf bf3 = load_bfrag(lB + ((p + 6) * 16 + ncol) * LDAB_ + c * 32 + seg * 16);
      acc[0] = wmma_bf16(af, bf0, acc[0]);
      acc[1] = wmma_bf16(af, bf1, acc[1]);
      acc[2] = wmma_bf16(af, bf2, acc[2]);
      acc[3] = wmma_bf16(af, bf3, acc[3]);
      SGB(0x100, 10, 0);   // 10 DS reads (A-frag 2 + B-frags 8)
      SGB(0x008, 4, 0);    // 4 WMMA
    }
    __syncthreads();
  }
}

// ---------------- Q/K projection + RoPE epilogue ----------------
// dst layout: [b][head][t][dh], head = blockIdx.y, #heads = gridDim.y
__global__ __launch_bounds__(512) void gemm_rope_kernel(const unsigned short* __restrict__ A,
                                                        const unsigned short* __restrict__ Bt,
                                                        unsigned short* __restrict__ dst,
                                                        float scale){
  __shared__ alignas(32) unsigned short lA[128 * LDAB_];
  __shared__ alignas(32) unsigned short lB[128 * LDAB_];
  v8f acc[4];
  gemm_core(A, Bt, C_, (size_t)blockIdx.x * 128, (size_t)blockIdx.y * 128, lA, lB, acc);

  const int lane = threadIdx.x & 31, wave = threadIdx.x >> 5;
  const int ncol = lane & 15, seg = lane >> 4;
  const int mt = wave >> 1, p = wave & 1;
  const int head = blockIdx.y;
  const int NH   = gridDim.y;
#pragma unroll
  for (int j = 0; j < 2; j++){
    const int d1 = (p + 2 * j) * 16 + ncol;               // 0..63 (tiles jj=p,p+2)
    const float invf = __expf((float)d1 * -0.14391156855509304f); // theta^(-d1/64)
#pragma unroll
    for (int r = 0; r < 8; r++){
      const int mg = blockIdx.x * 128 + mt * 16 + seg * 8 + r;   // row in [B*T]
      const int b = mg >> 11, t = mg & (T_ - 1);
      const float ang = (float)t * invf;
      const float cs = __cosf(ang), sn = __sinf(ang);
      const float x1 = acc[j][r], x2 = acc[j + 2][r];  // tile jj vs jj+4 (d1+64)
      const size_t base = (((size_t)b * NH + head) * T_ + t) * DH_;
      dst[base + d1]      = f2bf((x1 * cs - x2 * sn) * scale);
      dst[base + d1 + 64] = f2bf((x1 * sn + x2 * cs) * scale);
    }
  }
}

// ---------------- V projection, stored transposed: Vt[b][kv][d][t] ----------------
__global__ __launch_bounds__(512) void gemm_vT_kernel(const unsigned short* __restrict__ A,
                                                      const unsigned short* __restrict__ Bt,
                                                      unsigned short* __restrict__ dst){
  __shared__ alignas(32) unsigned short lA[128 * LDAB_];
  __shared__ alignas(32) unsigned short lB[128 * LDAB_];
  v8f acc[4];
  gemm_core(A, Bt, C_, (size_t)blockIdx.x * 128, (size_t)blockIdx.y * 128, lA, lB, acc);

  const int lane = threadIdx.x & 31, wave = threadIdx.x >> 5;
  const int ncol = lane & 15, seg = lane >> 4;
  const int mt = wave >> 1, p = wave & 1;
#pragma unroll
  for (int j = 0; j < 4; j++){
    const int d = (p + 2 * j) * 16 + ncol;
#pragma unroll
    for (int r = 0; r < 8; r++){
      const int mg = blockIdx.x * 128 + mt * 16 + seg * 8 + r;
      const int b = mg >> 11, t = mg & (T_ - 1);
      dst[(((size_t)b * HKV_ + blockIdx.y) * DH_ + d) * T_ + t] = f2bf(acc[j][r]);
    }
  }
}

// ---------------- output projection: fp32 result ----------------
__global__ __launch_bounds__(512) void gemm_out_kernel(const unsigned short* __restrict__ A,
                                                       const unsigned short* __restrict__ Bt,
                                                       float* __restrict__ out){
  __shared__ alignas(32) unsigned short lA[128 * LDAB_];
  __shared__ alignas(32) unsigned short lB[128 * LDAB_];
  v8f acc[4];
  gemm_core(A, Bt, C_, (size_t)blockIdx.x * 128, (size_t)blockIdx.y * 128, lA, lB, acc);

  const int lane = threadIdx.x & 31, wave = threadIdx.x >> 5;
  const int ncol = lane & 15, seg = lane >> 4;
  const int mt = wave >> 1, p = wave & 1;
#pragma unroll
  for (int j = 0; j < 4; j++){
#pragma unroll
    for (int r = 0; r < 8; r++){
      const int mg = blockIdx.x * 128 + mt * 16 + seg * 8 + r;
      out[(size_t)mg * C_ + blockIdx.y * 128 + (p + 2 * j) * 16 + ncol] = acc[j][r];
    }
  }
}

// ---------------- flash attention (causal, GQA) ----------------
// grid: (T/64, H, B); block 256 = 8 waves = 4 wave-pairs.
// Pair (wave>>1) owns one 16-row q-tile; both waves compute S redundantly
// (identical softmax state, no cross-wave sync); wave p=w&1 accumulates the
// 4 interleaved O column-tiles jd = p+2j.
__global__ __launch_bounds__(256) void attn_kernel(const unsigned short* __restrict__ Q,
                                                   const unsigned short* __restrict__ Kb,
                                                   const unsigned short* __restrict__ Vt,
                                                   unsigned short* __restrict__ Ao){
  __shared__ alignas(32) unsigned short lK[32 * LDK_];        // [s][d] padded
  __shared__ alignas(32) unsigned short lV[128 * LDV_];       // [d][s] padded
  __shared__ alignas(32) unsigned short lP[4][16 * LDP_];     // per-pair P tile [m][s] padded

  const int tid  = threadIdx.x;
  const int wave = tid >> 5, lane = tid & 31;
  const int ncol = lane & 15, seg = lane >> 4, koff = seg * 16;
  const int pair = wave >> 1, p = wave & 1;
  const int qt0 = blockIdx.x * 64;
  const int h = blockIdx.y, b = blockIdx.z, kv = h >> 2;
  const float NEGINF = -__builtin_inff();

  // per-lane Q row base (1/sqrt(dh) already folded into Q); re-read per tile (cache-hot)
  const unsigned short* gQ = Q + (((size_t)b * H_ + h) * T_ + qt0 + pair * 16 + ncol) * DH_;

  const v8f z = {0.f,0.f,0.f,0.f,0.f,0.f,0.f,0.f};
  v8f oacc[4];
#pragma unroll
  for (int j = 0; j < 4; j++) oacc[j] = z;
  // softmax running state as vectors (SSA values, never allocas)
  v8f mstv = {-3.0e38f,-3.0e38f,-3.0e38f,-3.0e38f,-3.0e38f,-3.0e38f,-3.0e38f,-3.0e38f};
  v8f lstv = z;

  // staging pointers (register double-buffer of the 32-key K/V tiles)
  const unsigned short* gKt = Kb + (((size_t)b * HKV_ + kv) * T_) * DH_ + tid * 16;
  const unsigned short* gVt = Vt + (((size_t)b * HKV_ + kv) * DH_ + (tid >> 1)) * T_ + (tid & 1) * 16;
  unsigned short* sK = lK + (tid >> 3) * LDK_ + (tid & 7) * 16;
  unsigned short* sV = lV + (tid >> 1) * LDV_ + (tid & 1) * 16;
  unsigned short* lPp = &lP[pair][0];

  uint4 rk0 = *(const uint4*)(gKt);
  uint4 rk1 = *(const uint4*)(gKt + 8);
  uint4 rv0 = *(const uint4*)(gVt);
  uint4 rv1 = *(const uint4*)(gVt + 8);

  const int send = qt0 + 64; // causal bound (uniform per block)
  for (int s0 = 0; s0 < send; s0 += 32){
    // commit staged K/V tile to LDS
    *(uint4*)sK       = rk0;  *(uint4*)(sK + 8) = rk1;
    *(uint4*)sV       = rv0;  *(uint4*)(sV + 8) = rv1;
    __syncthreads();
    // prefetch next tile into registers while we compute
    if (s0 + 32 < send){
      rk0 = *(const uint4*)(gKt + (size_t)(s0 + 32) * DH_);
      rk1 = *(const uint4*)(gKt + (size_t)(s0 + 32) * DH_ + 8);
      rv0 = *(const uint4*)(gVt + s0 + 32);
      rv1 = *(const uint4*)(gVt + s0 + 40);
    }

    // S = Q K^T (16 q-rows x 32 keys), duplicated across the wave pair
    v8f sa = z, sb = z;
#pragma unroll
    for (int c = 0; c < 4; c++){
      v16bf qfc = load_afrag(gQ + c * 32 + koff);
      v16bf k0f = load_bfrag(lK + (size_t)ncol * LDK_        + c * 32 + seg * 16);
      v16bf k1f = load_bfrag(lK + (size_t)(16 + ncol) * LDK_ + c * 32 + seg * 16);
      sa = wmma_bf16(qfc, k0f, sa);
      sb = wmma_bf16(qfc, k1f, sb);
    }

    // online softmax (identical in both waves of the pair)
#pragma unroll
    for (int r = 0; r < 8; r++){
      const int trow = qt0 + pair * 16 + seg * 8 + r;
      const float va = (s0 + ncol      <= trow) ? sa[r] : NEGINF;
      const float vb = (s0 + 16 + ncol <= trow) ? sb[r] : NEGINF;
      const float m_old = mstv[r];
      const float rm = rmax16(fmaxf(va, vb));
      const float mn = fmaxf(m_old, rm);
      const float alpha = __expf(m_old - mn);
      const float pa = __expf(va - mn);
      const float pb = __expf(vb - mn);
      lstv[r] = lstv[r] * alpha + rsum16(pa + pb);
      mstv[r] = mn;
#pragma unroll
      for (int j = 0; j < 4; j++) oacc[j][r] *= alpha;
      if (p == 0){
        const int mrow = seg * 8 + r;
        lPp[mrow * LDP_ + ncol]      = f2bf(pa);
        lPp[mrow * LDP_ + 16 + ncol] = f2bf(pb);
      }
    }
    __syncthreads();

    // O += P V : each wave covers its 4 interleaved d-tiles
    v16bf pf = load_afrag(lPp + ncol * LDP_ + koff);
    v16bf vf0 = load_bfrag(lV + (size_t)((p     ) * 16 + ncol) * LDV_ + seg * 16);
    v16bf vf1 = load_bfrag(lV + (size_t)((p + 2) * 16 + ncol) * LDV_ + seg * 16);
    v16bf vf2 = load_bfrag(lV + (size_t)((p + 4) * 16 + ncol) * LDV_ + seg * 16);
    v16bf vf3 = load_bfrag(lV + (size_t)((p + 6) * 16 + ncol) * LDV_ + seg * 16);
    oacc[0] = wmma_bf16(pf, vf0, oacc[0]);
    oacc[1] = wmma_bf16(pf, vf1, oacc[1]);
    oacc[2] = wmma_bf16(pf, vf2, oacc[2]);
    oacc[3] = wmma_bf16(pf, vf3, oacc[3]);
    SGB(0x100, 10, 0);   // 10 DS reads (P-frag 2 + V-frags 8)
    SGB(0x008, 4, 0);    // 4 WMMA
    __syncthreads();
  }

  // normalize + store attention output [b][t][h*128+d] as bf16
  v8f invv;
#pragma unroll
  for (int r = 0; r < 8; r++) invv[r] = 1.0f / lstv[r];
#pragma unroll
  for (int j = 0; j < 4; j++){
    const int d = (p + 2 * j) * 16 + ncol;
#pragma unroll
    for (int r = 0; r < 8; r++){
      const int t = qt0 + pair * 16 + seg * 8 + r;
      Ao[((size_t)b * T_ + t) * C_ + h * DH_ + d] = f2bf(oacc[j][r] * invv[r]);
    }
  }
}

// ---------------- launcher ----------------
extern "C" void kernel_launch(void* const* d_in, const int* in_sizes, int n_in,
                              void* d_out, int out_size, void* d_ws, size_t ws_size,
                              hipStream_t stream){
  (void)in_sizes; (void)n_in; (void)out_size; (void)ws_size;
  const float* X  = (const float*)d_in[0];
  const float* Wq = (const float*)d_in[1];
  const float* Wk = (const float*)d_in[2];
  const float* Wv = (const float*)d_in[3];
  const float* Wo = (const float*)d_in[4];

  // workspace layout (bytes); total 160 MiB. Abf reuses Xbf region.
  char* ws = (char*)d_ws;
  unsigned short* Xbf = (unsigned short*)(ws + 0ull);          // 32 MiB (X bf16, later attn-out)
  unsigned short* Wqt = (unsigned short*)(ws + 33554432ull);   // 32 MiB [4096][4096]
  unsigned short* Wkt = (unsigned short*)(ws + 67108864ull);   //  8 MiB [1024][4096]
  unsigned short* Wvt = (unsigned short*)(ws + 75497472ull);   //  8 MiB [1024][4096]
  unsigned short* Wot = (unsigned short*)(ws + 83886080ull);   // 32 MiB [4096][4096]
  unsigned short* Qbf = (unsigned short*)(ws + 117440512ull);  // 32 MiB [b][h][t][d]
  unsigned short* Kbf = (unsigned short*)(ws + 150994944ull);  //  8 MiB [b][kv][t][d]
  unsigned short* Vtb = (unsigned short*)(ws + 159383552ull);  //  8 MiB [b][kv][d][t]
  unsigned short* Abf = Xbf;

  // 1) precision conversion (+ weight transpose so GEMM B-frags are contiguous)
  cvt_bf16x4_kernel<<<(B_ * T_ * C_ / 4 + 255) / 256, 256, 0, stream>>>(X, Xbf, B_ * T_ * C_ / 4);
  transpose_cvt_kernel<<<dim3(C_ / 32, C_ / 32),   dim3(32, 8), 0, stream>>>(Wq, Wqt, C_, C_);
  transpose_cvt_kernel<<<dim3(1024 / 32, C_ / 32), dim3(32, 8), 0, stream>>>(Wk, Wkt, C_, 1024);
  transpose_cvt_kernel<<<dim3(1024 / 32, C_ / 32), dim3(32, 8), 0, stream>>>(Wv, Wvt, C_, 1024);
  transpose_cvt_kernel<<<dim3(C_ / 32, C_ / 32),   dim3(32, 8), 0, stream>>>(Wo, Wot, C_, C_);

  // 2) QKV projections (RoPE fused; 1/sqrt(dh) folded into Q; V stored transposed)
  const float qscale = 0.08838834764831845f; // 1/sqrt(128)
  gemm_rope_kernel<<<dim3((B_ * T_) / 128, H_),   512, 0, stream>>>(Xbf, Wqt, Qbf, qscale);
  gemm_rope_kernel<<<dim3((B_ * T_) / 128, HKV_), 512, 0, stream>>>(Xbf, Wkt, Kbf, 1.0f);
  gemm_vT_kernel  <<<dim3((B_ * T_) / 128, HKV_), 512, 0, stream>>>(Xbf, Wvt, Vtb);

  // 3) causal flash attention (GQA: 4 q-heads share each kv-head)
  attn_kernel<<<dim3(T_ / 64, H_, B_), 256, 0, stream>>>(Qbf, Kbf, Vtb, Abf);

  // 4) output projection -> fp32
  gemm_out_kernel<<<dim3((B_ * T_) / 128, C_ / 128), 512, 0, stream>>>(Abf, Wot, (float*)d_out);
}
